// PredictionModel_58841051955204
// MI455X (gfx1250) — matrix operand order repaired and checked by reading, not compiled, not run on hardware
//
#include <hip/hip_runtime.h>
#include <hip/hip_bf16.h>
#include <math.h>

// ---------------------------------------------------------------------------
// Model dims
// ---------------------------------------------------------------------------
#define BSZ   256
#define TT    100
#define NFEAT 64
#define VV    32
#define EE    8
#define HH    512
#define NCAT  16
#define INP   192           // IN_LEN=176 padded to multiple of 32
#define ROWS  (BSZ * TT)    // 25600
#define G4H   (4 * HH)      // 2048
#define NCONT 48
#define KTH   (HH / 32)     // 16 k-tiles for K=512

typedef __bf16 bf16;
typedef __attribute__((ext_vector_type(16))) __bf16 v16bf;
typedef __attribute__((ext_vector_type(8)))  __bf16 v8bf;
typedef __attribute__((ext_vector_type(8)))  float  v8f;

union afrag_u { v16bf v; v8bf h[2]; };

#define WMMA_BF16(A, B, C) \
    __builtin_amdgcn_wmma_f32_16x16x32_bf16(false, (A), false, (B), (short)0, (C), false, false)

// ---------------------------------------------------------------------------
// 1) Build padded bf16 feature matrix (ROWS x 192)
// ---------------------------------------------------------------------------
__global__ __launch_bounds__(256) void feat_kernel(
    const float* __restrict__ x, const float* __restrict__ mean,
    const float* __restrict__ scale, const float* __restrict__ emb,
    bf16* __restrict__ feat)
{
    int tid = blockIdx.x * blockDim.x + threadIdx.x;
    int row = tid >> 6;
    int k   = tid & 63;
    if (row >= ROWS) return;
    float xv = x[(size_t)row * NFEAT + k];
    int off = k + 7 * ((k + 3) >> 2);     // interleaved layout offset
    bf16* out = feat + (size_t)row * INP;
    if (k & 3) {
        out[off] = (bf16)((xv - mean[k]) / scale[k]);
    } else {
        int j   = k >> 2;
        int idx = (int)xv;
        idx = idx < 0 ? 0 : (idx > VV - 1 ? VV - 1 : idx);
        const float* e = emb + ((size_t)j * VV + idx) * EE;
        #pragma unroll
        for (int q = 0; q < EE; ++q) out[off + q] = (bf16)e[q];
    }
    if (k < INP - 176) out[176 + k] = (bf16)0.0f;
}

// ---------------------------------------------------------------------------
// 2) Pack f32 weights into bf16 WMMA B-fragment tile layout.
//    dst[((nt*KT + kt)*32 + lane)*16 + j] = B[kt*32 + (lane>>4)*8 + (j&7) + (j>>3)*16][nt*16 + (lane&15)]
//    mode 0: (K,N) row-major   mode 1: (N,K) transpose
//    mode 2: W_in (176,512) K-padded to 192   mode 3: W_cls (16,512,32)
// ---------------------------------------------------------------------------
__global__ __launch_bounds__(256) void pack_b_kernel(
    const float* __restrict__ src, bf16* __restrict__ dst,
    int K, int N, int mode)
{
    int tid = blockIdx.x * blockDim.x + threadIdx.x;
    if (tid >= K * N) return;
    int j    = tid & 15;
    int rest = tid >> 4;
    int lane = rest & 31;
    int tile = rest >> 5;
    int KT = K >> 5;
    int kt = tile % KT;
    int nt = tile / KT;
    int k = kt * 32 + ((lane >> 4) * 8) + (j & 7) + ((j >> 3) * 16);
    int n = nt * 16 + (lane & 15);
    float v;
    switch (mode) {
        case 0:  v = src[(size_t)k * N + n]; break;
        case 1:  v = src[(size_t)n * K + k]; break;
        case 2:  v = (k < 176) ? src[(size_t)k * 512 + n] : 0.0f; break;
        default: v = src[((size_t)(n >> 5) * 512 + k) * 32 + (n & 31)]; break;
    }
    dst[tid] = (bf16)v;
}

// ---------------------------------------------------------------------------
// 3a) Register-tiled WMMA GEMM: each wave computes 2(M) x 4(N) 16x16 tiles
//     -> 8 WMMAs per k-step from 6 fragment loads.
// ---------------------------------------------------------------------------
template <int ACT, bool OUTBF>
__global__ __launch_bounds__(256) void wmma_gemm_tiled_kernel(
    const bf16* __restrict__ A, int lda,
    const bf16* __restrict__ Bp, const float* __restrict__ bias,
    void* __restrict__ Cout, int ldc,
    int mtiles, int ntiles, int ktiles)
{
    int lane = threadIdx.x & 31;
    int wave = blockIdx.x * (blockDim.x >> 5) + (threadIdx.x >> 5);
    int mg = mtiles >> 1;
    int ng = ntiles >> 2;
    if (wave >= mg * ng) return;
    int mgi = wave / ng;
    int ngi = wave % ng;
    int mt0 = mgi * 2;
    int nt0 = ngi * 4;

    const bf16* arow0 = A + (size_t)(mt0 * 16 + (lane & 15)) * lda + ((lane >> 4) * 8);
    const bf16* arow1 = arow0 + (size_t)16 * lda;
    const bf16* bt0 = Bp + ((size_t)(nt0 + 0) * ktiles) * 512 + lane * 16;
    const bf16* bt1 = Bp + ((size_t)(nt0 + 1) * ktiles) * 512 + lane * 16;
    const bf16* bt2 = Bp + ((size_t)(nt0 + 2) * ktiles) * 512 + lane * 16;
    const bf16* bt3 = Bp + ((size_t)(nt0 + 3) * ktiles) * 512 + lane * 16;

    v8f acc[2][4] = {};
    for (int kt = 0; kt < ktiles; ++kt) {
        afrag_u a0, a1;
        a0.h[0] = *(const v8bf*)(arow0);
        a0.h[1] = *(const v8bf*)(arow0 + 16);
        a1.h[0] = *(const v8bf*)(arow1);
        a1.h[1] = *(const v8bf*)(arow1 + 16);
        v16bf b0 = *(const v16bf*)(bt0);
        v16bf b1 = *(const v16bf*)(bt1);
        v16bf b2 = *(const v16bf*)(bt2);
        v16bf b3 = *(const v16bf*)(bt3);
        acc[0][0] = WMMA_BF16(a0.v, b0, acc[0][0]);
        acc[0][1] = WMMA_BF16(a0.v, b1, acc[0][1]);
        acc[0][2] = WMMA_BF16(a0.v, b2, acc[0][2]);
        acc[0][3] = WMMA_BF16(a0.v, b3, acc[0][3]);
        acc[1][0] = WMMA_BF16(a1.v, b0, acc[1][0]);
        acc[1][1] = WMMA_BF16(a1.v, b1, acc[1][1]);
        acc[1][2] = WMMA_BF16(a1.v, b2, acc[1][2]);
        acc[1][3] = WMMA_BF16(a1.v, b3, acc[1][3]);
        arow0 += 32; arow1 += 32;
        bt0 += 512; bt1 += 512; bt2 += 512; bt3 += 512;
    }

    #pragma unroll
    for (int ni = 0; ni < 4; ++ni) {
        int n = (nt0 + ni) * 16 + (lane & 15);
        float bv = bias ? bias[n] : 0.0f;
        #pragma unroll
        for (int mi = 0; mi < 2; ++mi) {
            #pragma unroll
            for (int r = 0; r < 8; ++r) {
                int m = (mt0 + mi) * 16 + r + 8 * (lane >> 4);
                float v = acc[mi][ni][r] + bv;
                if (ACT == 1) v = v > 0.0f ? v : 0.0f;
                if (OUTBF) ((bf16*) Cout)[(size_t)m * ldc + n] = (bf16)v;
                else       ((float*)Cout)[(size_t)m * ldc + n] = v;
            }
        }
    }
}

// ---------------------------------------------------------------------------
// 3b) 1x1-tile fallback GEMM (small heads with ntiles not % 4)
// ---------------------------------------------------------------------------
template <int ACT, bool OUTBF>
__global__ __launch_bounds__(256) void wmma_gemm_kernel(
    const bf16* __restrict__ A, int lda,
    const bf16* __restrict__ Bp, const float* __restrict__ bias,
    void* __restrict__ Cout, int ldc,
    int mtiles, int ntiles, int ktiles)
{
    int lane = threadIdx.x & 31;
    int wave = blockIdx.x * (blockDim.x >> 5) + (threadIdx.x >> 5);
    if (wave >= mtiles * ntiles) return;
    int mt = wave / ntiles;
    int nt = wave % ntiles;

    const bf16* arow  = A + (size_t)(mt * 16 + (lane & 15)) * lda + ((lane >> 4) * 8);
    const bf16* btile = Bp + ((size_t)nt * ktiles) * 512 + lane * 16;

    v8f acc = {};
    for (int kt = 0; kt < ktiles; ++kt) {
        afrag_u ua;
        ua.h[0] = *(const v8bf*)(arow);
        ua.h[1] = *(const v8bf*)(arow + 16);
        v16bf b = *(const v16bf*)(btile);
        acc = WMMA_BF16(ua.v, b, acc);
        arow  += 32;
        btile += 512;
    }
    int n = nt * 16 + (lane & 15);
    float bv = bias ? bias[n] : 0.0f;
    #pragma unroll
    for (int r = 0; r < 8; ++r) {
        int m = mt * 16 + r + 8 * (lane >> 4);
        float v = acc[r] + bv;
        if (ACT == 1) v = v > 0.0f ? v : 0.0f;
        if (OUTBF) ((bf16*) Cout)[(size_t)m * ldc + n] = (bf16)v;
        else       ((float*)Cout)[(size_t)m * ldc + n] = v;
    }
}

// ---------------------------------------------------------------------------
// 4) Persistent LSTM layer kernel: one launch runs all T=100 steps.
//    - 32 blocks (one hidden tile each) x 8 waves (2 batch tiles each)
//    - Whh gate tiles (64 KB) staged once into LDS -> per-step B via ds_load
//    - cell state c kept in registers for the whole sequence
//    - steps separated by an agent-scope atomic grid barrier
// ---------------------------------------------------------------------------
__device__ __forceinline__ float sigmoidf_(float x) { return 1.0f / (1.0f + __expf(-x)); }

__global__ __launch_bounds__(256) void lstm_persist_kernel(
    const bf16* __restrict__ Whhp,     // packed B tiles (KT=16, N=2048)
    const bf16* __restrict__ xw,       // (ROWS, 2048) bf16, rows are (b*T+t)
    const float* __restrict__ bhh,     // (2048,)
    bf16* __restrict__ hrecA,          // (256,512) bf16, zeroed (t=0 input)
    bf16* __restrict__ hrecB,          // (256,512) bf16
    bf16* __restrict__ hseq_out,       // (ROWS,512) bf16  <- relu(h)
    unsigned* __restrict__ syncc)      // zeroed grid-barrier counter
{
    __shared__ __align__(16) bf16 smem[4 * KTH * 512];   // 64 KB: 4 gate tile-columns

    int lane = threadIdx.x & 31;
    int mgi  = threadIdx.x >> 5;       // wave in block = batch-tile group (0..7)
    int ht   = blockIdx.x;             // hidden tile (0..31)
    int mt0  = mgi * 2;

    // --- stage this block's Whh gate tiles into LDS (4 contiguous 16KB runs)
    #pragma unroll
    for (int g = 0; g < 4; ++g) {
        const uint4* src = (const uint4*)(Whhp + ((size_t)(g * 32 + ht) * KTH) * 512);
        uint4* dst = (uint4*)(smem + (size_t)g * KTH * 512);
        for (int i = threadIdx.x; i < (KTH * 512) / 8; i += 256) dst[i] = src[i];
    }
    __syncthreads();

    int n = ht * 16 + (lane & 15);
    float bi = bhh[n], bfv = bhh[HH + n], bgv = bhh[2 * HH + n], bov = bhh[3 * HH + n];

    float cfr[2][8];
    #pragma unroll
    for (int mi = 0; mi < 2; ++mi)
        #pragma unroll
        for (int r = 0; r < 8; ++r) cfr[mi][r] = 0.0f;

    for (int t = 0; t < TT; ++t) {
        const bf16* hin  = (t & 1) ? hrecB : hrecA;
        bf16*       hout = (t & 1) ? hrecA : hrecB;

        const bf16* arow0 = hin + (size_t)(mt0 * 16 + (lane & 15)) * HH + ((lane >> 4) * 8);
        const bf16* arow1 = arow0 + (size_t)16 * HH;
        const bf16* sb0 = smem + (size_t)(0 * KTH) * 512 + lane * 16;
        const bf16* sb1 = smem + (size_t)(1 * KTH) * 512 + lane * 16;
        const bf16* sb2 = smem + (size_t)(2 * KTH) * 512 + lane * 16;
        const bf16* sb3 = smem + (size_t)(3 * KTH) * 512 + lane * 16;

        v8f acc[2][4] = {};
        for (int kt = 0; kt < KTH; ++kt) {
            afrag_u a0, a1;
            a0.h[0] = *(const v8bf*)(arow0);
            a0.h[1] = *(const v8bf*)(arow0 + 16);
            a1.h[0] = *(const v8bf*)(arow1);
            a1.h[1] = *(const v8bf*)(arow1 + 16);
            v16bf b0 = *(const v16bf*)(sb0);
            v16bf b1 = *(const v16bf*)(sb1);
            v16bf b2 = *(const v16bf*)(sb2);
            v16bf b3 = *(const v16bf*)(sb3);
            acc[0][0] = WMMA_BF16(a0.v, b0, acc[0][0]);
            acc[0][1] = WMMA_BF16(a0.v, b1, acc[0][1]);
            acc[0][2] = WMMA_BF16(a0.v, b2, acc[0][2]);
            acc[0][3] = WMMA_BF16(a0.v, b3, acc[0][3]);
            acc[1][0] = WMMA_BF16(a1.v, b0, acc[1][0]);
            acc[1][1] = WMMA_BF16(a1.v, b1, acc[1][1]);
            acc[1][2] = WMMA_BF16(a1.v, b2, acc[1][2]);
            acc[1][3] = WMMA_BF16(a1.v, b3, acc[1][3]);
            arow0 += 32; arow1 += 32;
            sb0 += 512; sb1 += 512; sb2 += 512; sb3 += 512;
        }

        #pragma unroll
        for (int mi = 0; mi < 2; ++mi) {
            #pragma unroll
            for (int r = 0; r < 8; ++r) {
                int m = (mt0 + mi) * 16 + r + 8 * (lane >> 4);       // batch index
                size_t xb = ((size_t)m * TT + t) * G4H + n;
                float pi = acc[mi][0][r] + (float)xw[xb]          + bi;
                float pf = acc[mi][1][r] + (float)xw[xb + HH]     + bfv;
                float pg = acc[mi][2][r] + (float)xw[xb + 2 * HH] + bgv;
                float po = acc[mi][3][r] + (float)xw[xb + 3 * HH] + bov;
                float ig = sigmoidf_(pi);
                float fg = sigmoidf_(pf);
                float gg = tanhf(pg);
                float og = sigmoidf_(po);
                float cn = fg * cfr[mi][r] + ig * gg;
                float hv = og * tanhf(cn);
                cfr[mi][r] = cn;
                hout[(size_t)m * HH + n] = (bf16)hv;
                hseq_out[((size_t)m * TT + t) * HH + n] = (bf16)(hv > 0.0f ? hv : 0.0f);
            }
        }

        if (t != TT - 1) {
            // grid barrier: release my h stores, count all blocks, acquire theirs
            __threadfence();
            __syncthreads();
            if (threadIdx.x == 0) {
                __hip_atomic_fetch_add(syncc, 1u, __ATOMIC_ACQ_REL, __HIP_MEMORY_SCOPE_AGENT);
                unsigned target = (unsigned)(t + 1) * (unsigned)gridDim.x;
                while (__hip_atomic_load(syncc, __ATOMIC_ACQUIRE, __HIP_MEMORY_SCOPE_AGENT) < target)
                    __builtin_amdgcn_s_sleep(1);
            }
            __syncthreads();
            __threadfence();
        }
    }
}

// ---------------------------------------------------------------------------
// 5) zero recurrent state + barrier counter
// ---------------------------------------------------------------------------
__global__ __launch_bounds__(256) void init_state_kernel(bf16* h0, bf16* h1, unsigned* syncc)
{
    int i = blockIdx.x * blockDim.x + threadIdx.x;
    if (i < BSZ * HH) { h0[i] = (bf16)0.0f; h1[i] = (bf16)0.0f; }
    if (i == 0) *syncc = 0u;
}

// ---------------------------------------------------------------------------
// launch
// ---------------------------------------------------------------------------
static inline size_t alignup(size_t x) { return (x + 255) & ~(size_t)255; }

extern "C" void kernel_launch(void* const* d_in, const int* in_sizes, int n_in,
                              void* d_out, int out_size, void* d_ws, size_t ws_size,
                              hipStream_t stream)
{
    const float* x      = (const float*)d_in[0];
    const float* smean  = (const float*)d_in[1];
    const float* sscale = (const float*)d_in[2];
    const float* emb    = (const float*)d_in[3];
    const float* W_in   = (const float*)d_in[4];
    const float* b_in   = (const float*)d_in[5];
    const float* Wih    = (const float*)d_in[6];   // (2, 2048, 512)
    const float* Whh    = (const float*)d_in[7];   // (2, 2048, 512)
    const float* bih    = (const float*)d_in[8];   // (2, 2048)
    const float* bhh    = (const float*)d_in[9];   // (2, 2048)
    const float* W_out  = (const float*)d_in[10];  // (512, 48)
    const float* b_out  = (const float*)d_in[11];  // (48,)
    const float* W_cls  = (const float*)d_in[12];  // (16, 512, 32)
    const float* b_cls  = (const float*)d_in[13];  // (16, 32) -> flat 512
    float* out = (float*)d_out;

    // workspace layout
    char* base = (char*)d_ws;
    size_t off = 0;
    auto take = [&](size_t bytes) { char* p = base + off; off += alignup(bytes); return p; };
    bf16* feat   = (bf16*)take((size_t)ROWS * INP * 2);
    bf16* bufH0  = (bf16*)take((size_t)ROWS * HH * 2);
    bf16* bufH1  = (bf16*)take((size_t)ROWS * HH * 2);
    bf16* xw     = (bf16*)take((size_t)ROWS * G4H * 2);
    bf16* WinP   = (bf16*)take((size_t)INP * HH * 2);
    bf16* WihP0  = (bf16*)take((size_t)HH * G4H * 2);
    bf16* WihP1  = (bf16*)take((size_t)HH * G4H * 2);
    bf16* WhhP0  = (bf16*)take((size_t)HH * G4H * 2);
    bf16* WhhP1  = (bf16*)take((size_t)HH * G4H * 2);
    bf16* WoutP  = (bf16*)take((size_t)HH * NCONT * 2);
    bf16* WclsP  = (bf16*)take((size_t)HH * (NCAT * VV) * 2);
    bf16* hrecA  = (bf16*)take((size_t)BSZ * HH * 2);
    bf16* hrecB  = (bf16*)take((size_t)BSZ * HH * 2);
    unsigned* syncc = (unsigned*)take(256);
    (void)ws_size; (void)in_sizes; (void)n_in; (void)out_size;

    // --- features + weight packing ---
    feat_kernel<<<(ROWS * NFEAT + 255) / 256, 256, 0, stream>>>(x, smean, sscale, emb, feat);
    pack_b_kernel<<<(INP * HH + 255) / 256, 256, 0, stream>>>(W_in, WinP, INP, HH, 2);
    pack_b_kernel<<<(HH * G4H + 255) / 256, 256, 0, stream>>>(Wih,                  WihP0, HH, G4H, 1);
    pack_b_kernel<<<(HH * G4H + 255) / 256, 256, 0, stream>>>(Wih + (size_t)G4H*HH, WihP1, HH, G4H, 1);
    pack_b_kernel<<<(HH * G4H + 255) / 256, 256, 0, stream>>>(Whh,                  WhhP0, HH, G4H, 1);
    pack_b_kernel<<<(HH * G4H + 255) / 256, 256, 0, stream>>>(Whh + (size_t)G4H*HH, WhhP1, HH, G4H, 1);
    pack_b_kernel<<<(HH * NCONT + 255) / 256, 256, 0, stream>>>(W_out, WoutP, HH, NCONT, 0);
    pack_b_kernel<<<(HH * NCAT * VV + 255) / 256, 256, 0, stream>>>(W_cls, WclsP, HH, NCAT * VV, 3);

    // --- input projection: bufH0 = relu(feat @ W_in + b_in) ---
    {
        int mt = ROWS / 16, nt = HH / 16, kt = INP / 32;      // 1600, 32, 6
        int waves = (mt / 2) * (nt / 4);                      // 6400
        wmma_gemm_tiled_kernel<1, true><<<(waves + 7) / 8, 256, 0, stream>>>(
            feat, INP, WinP, b_in, bufH0, HH, mt, nt, kt);
    }

    // --- 2 LSTM layers ---
    for (int l = 0; l < 2; ++l) {
        const bf16* hin  = (l == 0) ? bufH0 : bufH1;
        bf16*       hout = (l == 0) ? bufH1 : bufH0;
        const bf16* WihP = (l == 0) ? WihP0 : WihP1;
        const bf16* WhhP = (l == 0) ? WhhP0 : WhhP1;

        // big parallel GEMM: xw = hin @ Wihᵀ + bih   (25600 x 2048, K=512)
        {
            int mt = ROWS / 16, nt = G4H / 16, kt = HH / 32;  // 1600, 128, 16
            int waves = (mt / 2) * (nt / 4);                  // 25600
            wmma_gemm_tiled_kernel<0, true><<<(waves + 7) / 8, 256, 0, stream>>>(
                hin, HH, WihP, bih + (size_t)l * G4H, xw, G4H, mt, nt, kt);
        }
        init_state_kernel<<<(BSZ * HH + 255) / 256, 256, 0, stream>>>(hrecA, hrecB, syncc);

        // single persistent launch runs the whole T=100 recurrence
        lstm_persist_kernel<<<32, 256, 0, stream>>>(
            WhhP, xw, bhh + (size_t)l * G4H, hrecA, hrecB, hout, syncc);
    }

    // --- heads on last timestep (layer-1 output lives in bufH0) ---
    const bf16* last = bufH0 + (size_t)(TT - 1) * HH;   // row b at stride T*H
    {
        int mt = BSZ / 16, nt = NCONT / 16, kt = HH / 32;     // 16, 3, 16
        wmma_gemm_kernel<0, false><<<(mt * nt + 7) / 8, 256, 0, stream>>>(
            last, TT * HH, WoutP, b_out, out, NCONT, mt, nt, kt);
    }
    {
        int mt = BSZ / 16, nt = (NCAT * VV) / 16, kt = HH / 32;  // 16, 32, 16
        int waves = (mt / 2) * (nt / 4);                         // 64
        wmma_gemm_tiled_kernel<0, false><<<(waves + 7) / 8, 256, 0, stream>>>(
            last, TT * HH, WclsP, b_cls, out + (size_t)BSZ * NCONT, NCAT * VV, mt, nt, kt);
    }
}
